// SPVCNN_20615843021045
// MI455X (gfx1250) — compile-verified, matrix operand order checked by reading, not run on hardware
//
#include <hip/hip_runtime.h>

#define EPSV 1e-5f

// ---------------------------------------------------------------------------
// CDNA5 WMMA types
// ---------------------------------------------------------------------------
typedef __attribute__((ext_vector_type(16))) __bf16 v16bf;
typedef __attribute__((ext_vector_type(8)))  float  v8f;

union BFrag {
  v16bf v;
  unsigned short u[16];
  unsigned int   x[8];
};

__device__ __forceinline__ unsigned short f2bf(float f) {
  union { float f; unsigned u; } c; c.f = f;
  unsigned r = c.u + 0x7fffu + ((c.u >> 16) & 1u);   // round-to-nearest-even
  return (unsigned short)(r >> 16);
}

// Fragment-ready layout (CDNA5 16-bit operand layout, wave32):
//   consumer lane L (nn=L&15, hi=L>>4), element e (0..15):
//     p=e>>1, off=e&1, kb = (p<4 ? 8*hi+2p : 16+8*hi+2(p-4)) + off
//   inverse: element (row, kk) lives at lane = row | (((kk>>3)&1)<<4),
//            e = (kk&7) | ((kk&16)>>1)

// ---------------------------------------------------------------------------
// Weight pre-swizzle: W[k][ci][co] (f32) -> bf16 tiles of 512 entries, one per
// (k, ci-chunk-of-32, co-tile-of-16), each stored in exact per-lane fragment
// order so gg_wmma reads B as two coalesced b128 loads per lane.
// ---------------------------------------------------------------------------
__global__ void k_swizzle_w(const float* __restrict__ W,
                            unsigned short* __restrict__ o,
                            int K, int cin, int cout) {
  const int tile = blockIdx.x;
  const int cch = (cin + 31) >> 5;
  const int coT = (cout + 15) >> 4;
  const int k   = tile / (cch * coT);
  const int rem = tile - k * cch * coT;
  const int cc  = rem / coT;
  const int ct  = rem - cc * coT;
  for (int t = threadIdx.x; t < 512; t += blockDim.x) {
    const int lane = t >> 4, e = t & 15;
    const int nn = lane & 15, hi = lane >> 4;
    const int p = e >> 1, off = e & 1;
    const int kb = ((p < 4) ? (8 * hi + 2 * p) : (16 + 8 * hi + 2 * (p - 4))) + off;
    const int ci = (cc << 5) + kb;
    const int co = (ct << 4) + nn;
    const float v = (ci < cin && co < cout)
                      ? W[((size_t)k * cin + ci) * cout + co] : 0.f;
    o[(size_t)tile * 512 + t] = f2bf(v);
  }
}

// ---------------------------------------------------------------------------
// Fused gather-GEMM + bias + BN + residual + ReLU.
//   out[m,co] = act( (sum_{k,ci} A[m,k,ci]*W[k,ci,co] + bias)*bn_sc + bn_sh
//                    + res[m,co] )
//   A[m,k,ci] = (idx ? (idx[m,k]>=0 ? X[idx[m,k],ci] : 0) : X[m,ci])
// Block = 4 waves = 16 rows x 64 cols.  A tile gathered once per chunk into
// fragment-ready LDS (double buffered, 1 barrier/chunk), shared by 4 waves.
// B read directly from pre-swizzled global bf16 (L2-resident).
// ALIGNED = (cin % 32 == 0): A row loads are one b128, masking is a 64-bit
// AND of the packed bf16 (cannot be re-predicated by the compiler).
// ---------------------------------------------------------------------------
template <bool ALIGNED>
__global__ __launch_bounds__(128) void gg_wmma(
    const float* __restrict__ X, int cin,
    const int*   __restrict__ idx, int K,
    const unsigned short* __restrict__ Wsw,
    const float* __restrict__ bias,
    const float* __restrict__ bng, const float* __restrict__ bnb,
    const float* __restrict__ bnm, const float* __restrict__ bnv,
    const float* __restrict__ res, int res_ld,
    float* __restrict__ out, int out_ld,
    int M, int cout, int relu)
{
  __shared__ __align__(16) unsigned short lA[2][512];   // fragment-ready A

  const int tid  = threadIdx.x;
  const int wave = tid >> 5;
  const int lane = tid & 31;
  const int m0   = blockIdx.x * 16;
  const int nn   = lane & 15;

  const int cch     = (cin + 31) >> 5;
  const int coTiles = (cout + 15) >> 4;
  const int ctRaw   = blockIdx.y * 4 + wave;
  const int ctC     = (ctRaw < coTiles) ? ctRaw : (coTiles - 1); // clamped reads
  const int co0     = ctRaw << 4;                                // real columns

  // A-loader assignment: thread loads 4 cols of one row per chunk.
  const int ar  = tid >> 3;            // row 0..15
  const int ac0 = (tid & 7) << 2;      // col base 0,4,...,28
  const int tl  = ar | (((ac0 >> 3) & 1) << 4);
  const int eb  = (ac0 & 7) | ((ac0 & 16) >> 1);

  v8f acc = {0.f, 0.f, 0.f, 0.f, 0.f, 0.f, 0.f, 0.f};
  int buf = 0;

  for (int k = 0; k < K; ++k) {
    // gather row index once per k
    const int m = m0 + ar;
    int g = -1;
    if (m < M) g = idx ? idx[(size_t)m * K + k] : m;
    const float* xr = X + (size_t)(g < 0 ? 0 : g) * cin;
    const unsigned long long rmask = (g >= 0) ? ~0ull : 0ull;

    for (int cc = 0; cc < cch; ++cc) {
      // ---- stage A tile (unconditional loads; zero-mask via integer AND) ---
      {
        const int cb = (cc << 5) + ac0;
        unsigned long long pk;
        if (ALIGNED) {
          const float4 fv = *(const float4*)(xr + cb);   // one b128 load
          pk =  (unsigned long long)f2bf(fv.x)
             | ((unsigned long long)f2bf(fv.y) << 16)
             | ((unsigned long long)f2bf(fv.z) << 32)
             | ((unsigned long long)f2bf(fv.w) << 48);
          pk &= rmask;
        } else {
          pk = 0;
          #pragma unroll
          for (int j = 0; j < 4; ++j) {
            const int ci  = cb + j;
            const int cic = (ci < cin) ? ci : 0;
            const unsigned long long em =
                (ci < cin) ? 0xffffull : 0ull;           // element mask
            pk |= ((unsigned long long)f2bf(xr[cic]) & em) << (16 * j);
          }
          pk &= rmask;
        }
        *(unsigned long long*)(&lA[buf][tl * 16 + eb]) = pk;
      }
      __syncthreads();

      // ---- fragments: A from LDS (2 x b128), B from global (2 x b128) ----
      BFrag a, b;
      const unsigned* pa = (const unsigned*)(&lA[buf][lane << 4]);
      const unsigned* pb = (const unsigned*)(
          Wsw + ((size_t)((k * cch + cc) * coTiles + ctC) << 9) + (lane << 4));
      #pragma unroll
      for (int i = 0; i < 8; ++i) { a.x[i] = pa[i]; b.x[i] = pb[i]; }

      acc = __builtin_amdgcn_wmma_f32_16x16x32_bf16(
                false, a.v, false, b.v, (short)0, acc, false, false);
      buf ^= 1;
    }
  }

  // ---- epilogue: bias -> BN -> residual -> ReLU -> store ----
  const int hi = lane >> 4;
  const int co = co0 + nn;
  if (co < cout) {
    float sc = 1.f, sh = 0.f;
    if (bng) {
      sc = bng[co] * rsqrtf(bnv[co] + EPSV);
      sh = bnb[co] - bnm[co] * sc;
    }
    const float bs = bias ? bias[co] : 0.f;
    #pragma unroll
    for (int j = 0; j < 8; ++j) {
      const int m = m0 + j + (hi << 3);
      if (m < M) {
        float v = (acc[j] + bs) * sc + sh;
        if (res)  v += res[(size_t)m * res_ld + co];
        if (relu) v = fmaxf(v, 0.f);
        out[(size_t)m * out_ld + co] = v;
      }
    }
  }
}

// ---------------------------------------------------------------------------
// Elementwise / scatter-gather helpers (2D grids: no div/mod per thread)
// ---------------------------------------------------------------------------
__global__ void k_fill0(float* p, long long n) {
  long long i = (long long)blockIdx.x * blockDim.x + threadIdx.x;
  if (i < n) p[i] = 0.f;
}

__global__ void k_p2v_scatter(const float* __restrict__ z,
                              const int* __restrict__ idx,
                              float* __restrict__ vox, int C) {
  const int c0 = blockIdx.x * blockDim.x + threadIdx.x;
  const int n  = blockIdx.y;
  if (c0 < C) atomicAdd(vox + (size_t)idx[n] * C + c0, z[(size_t)n * C + c0]);
}

__global__ void k_rows_scale(float* vox, const float* __restrict__ cnt, int C) {
  const int c0 = blockIdx.x * blockDim.x + threadIdx.x;
  const int n  = blockIdx.y;
  if (c0 < C) vox[(size_t)n * C + c0] *= (1.f / cnt[n]);
}

__global__ void k_v2p(const float* __restrict__ vox,
                      const int* __restrict__ idx8,
                      const float* __restrict__ w8,
                      const float* __restrict__ add,
                      float* __restrict__ out, int C) {
  const int c0 = blockIdx.x * blockDim.x + threadIdx.x;
  const int n  = blockIdx.y;
  if (c0 >= C) return;
  float s = add ? add[(size_t)n * C + c0] : 0.f;
  #pragma unroll
  for (int k = 0; k < 8; ++k) {
    const int g  = idx8[n * 8 + k];
    const int gc = (g < 0) ? 0 : g;
    float wv = w8[n * 8 + k];
    if (g < 0) wv = 0.f;                       // mask the weight, not the load
    s += vox[(size_t)gc * C + c0] * wv;
  }
  out[(size_t)n * C + c0] = s;
}

__global__ void k_copy_cols(const float* __restrict__ src, int sld,
                            float* __restrict__ dst, int dld, int cols) {
  const int c0 = blockIdx.x * blockDim.x + threadIdx.x;
  const int r  = blockIdx.y;
  if (c0 < cols) dst[(size_t)r * dld + c0] = src[(size_t)r * sld + c0];
}

// Expand tconv (qidx, kidx) into a masked K=8 gather table.
__global__ void k_build_tidx(const int* __restrict__ q,
                             const int* __restrict__ kk,
                             int* __restrict__ t, int M) {
  int i = blockIdx.x * blockDim.x + threadIdx.x;
  if (i >= M) return;
  const int qi = q[i], kv = kk[i];
  #pragma unroll
  for (int k = 0; k < 8; ++k) t[i * 8 + k] = (k == kv && qi >= 0) ? qi : -1;
}

// ---------------------------------------------------------------------------
// Host-side parameter bundles (flattened jax pytree, sorted dict keys)
// ---------------------------------------------------------------------------
struct BNp  { const float *b, *g, *m, *v; };
struct RBp  { BNp bn1, bn2, bnd; const unsigned short *w1, *w2, *wd; };
struct Linp { const float* b; BNp bn; const unsigned short* w; };
struct Stp  { BNp dbn; const unsigned short* dw; RBp r0, r1; };
struct Upp  { BNp bn; RBp r0, r1; const unsigned short* w; };

struct Bump {
  char*  base;
  size_t off;
  void* take(size_t bytes) {
    size_t a = (off + 255) & ~(size_t)255;
    off = a + bytes;
    return base + a;
  }
};

extern "C" void kernel_launch(void* const* d_in, const int* in_sizes, int n_in,
                              void* d_out, int out_size, void* d_ws, size_t ws_size,
                              hipStream_t stream) {
  (void)n_in; (void)out_size; (void)ws_size;

  auto F = [&](int i) { return (const float*)d_in[i]; };
  auto I = [&](int i) { return (const int*)d_in[i]; };

  // Top-level pytree order: sorted -> feat, maps(29), params(265).
  // If harness kept insertion order (feat, params, maps), detect via cls_b.
  int mbase, pbase;
  if (in_sizes[1] == 19) { pbase = 1; mbase = 1 + 265; }
  else                   { mbase = 1; pbase = 1 + 29; }

  const int N  = in_sizes[0] / 4;
  const int n1 = in_sizes[mbase + 0] / 8;
  const int n2 = in_sizes[mbase + 1] / 8;
  const int n3 = in_sizes[mbase + 2] / 8;
  const int n4 = in_sizes[mbase + 3] / 8;
  const int n0 = in_sizes[mbase + 10] / 27;

  const float* feat = F(0);
  const int* down1 = I(mbase + 0);  const int* down2 = I(mbase + 1);
  const int* down3 = I(mbase + 2);  const int* down4 = I(mbase + 3);
  const int* dv0_idx = I(mbase + 4);  const float* dv0_w = F(mbase + 5);
  const int* dv2_idx = I(mbase + 6);  const float* dv2_w = F(mbase + 7);
  const int* dv4_idx = I(mbase + 8);  const float* dv4_w = F(mbase + 9);
  const int* nbr0 = I(mbase + 10); const int* nbr1 = I(mbase + 11);
  const int* nbr2 = I(mbase + 12); const int* nbr3 = I(mbase + 13);
  const int* nbr4 = I(mbase + 14);
  const float* pv0_cnt = F(mbase + 15); const int* pv0_idx = I(mbase + 16);
  const float* pv2_cnt = F(mbase + 17); const int* pv2_idx = I(mbase + 18);
  const float* pv4_cnt = F(mbase + 19); const int* pv4_idx = I(mbase + 20);
  const int* up0_k = I(mbase + 21); const int* up0_q = I(mbase + 22);
  const int* up1_k = I(mbase + 23); const int* up1_q = I(mbase + 24);
  const int* up2_k = I(mbase + 25); const int* up2_q = I(mbase + 26);
  const int* up3_k = I(mbase + 27); const int* up3_q = I(mbase + 28);

  // ------------------------- workspace allocator ---------------------------
  Bump P{ (char*)d_ws, 0 };
  auto fbuf = [&](Bump& bp, size_t n) { return (float*)bp.take(n * sizeof(float)); };
  auto ibuf = [&](Bump& bp, size_t n) { return (int*)bp.take(n * sizeof(int)); };

  // Pre-swizzle a weight tensor into fragment-ready bf16 tiles.
  auto swz = [&](const float* W, int K, int cin, int cout) -> const unsigned short* {
    const int cch = (cin + 31) >> 5, coT = (cout + 15) >> 4;
    const int tiles = K * cch * coT;
    unsigned short* o = (unsigned short*)P.take((size_t)tiles * 512 * sizeof(unsigned short));
    k_swizzle_w<<<tiles, 256, 0, stream>>>(W, o, K, cin, cout);
    return o;
  };

  // Params, alphabetically sorted leaves, swizzled on the fly.
  int c = pbase;
  auto rdBN = [&]() { BNp x; x.b = F(c++); x.g = F(c++); x.m = F(c++); x.v = F(c++); return x; };
  auto rdRB = [&](bool wd, int ci, int co) {
    RBp r{}; r.bn1 = rdBN(); r.bn2 = rdBN();
    if (wd) r.bnd = rdBN();
    r.w1 = swz(F(c++), 27, ci, co);
    r.w2 = swz(F(c++), 27, co, co);
    r.wd = wd ? swz(F(c++), 1, ci, co) : nullptr;
    return r;
  };
  auto rdLin = [&](int ci, int co) {
    Linp l; l.b = F(c++); l.bn = rdBN(); l.w = swz(F(c++), 1, ci, co); return l;
  };
  auto rdStage = [&](bool wd0, int ci, int co) {
    Stp s; s.dbn = rdBN(); s.dw = swz(F(c++), 8, ci, ci);
    s.r0 = rdRB(wd0, ci, co); s.r1 = rdRB(false, co, co); return s;
  };
  auto rdUp = [&](int cx, int cy, int cskip) {
    Upp u; u.bn = rdBN();
    u.r0 = rdRB(true, cy + cskip, cy); u.r1 = rdRB(false, cy, cy);
    u.w = swz(F(c++), 8, cx, cy); return u;
  };

  const float* cls_b = F(c++);
  const unsigned short* cls_w = swz(F(c++), 1, 96, 19);
  Linp pt0 = rdLin(32, 256), pt1 = rdLin(256, 128), pt2 = rdLin(128, 96);
  Stp st1 = rdStage(false, 32, 32), st2 = rdStage(true, 32, 64),
      st3 = rdStage(true, 64, 128), st4 = rdStage(true, 128, 256);
  BNp stem_bn1 = rdBN(), stem_bn2 = rdBN();
  const unsigned short* stem_w1 = swz(F(c++), 27, 4, 32);
  const unsigned short* stem_w2 = swz(F(c++), 27, 32, 32);
  Upp u1 = rdUp(256, 256, 128), u2 = rdUp(256, 128, 64),
      u3 = rdUp(128, 96, 32),   u4 = rdUp(96, 96, 32);

  // ------------------------- activation buffers ----------------------------
  float* X0 = fbuf(P, (size_t)n0 * 32);
  float* X1 = fbuf(P, (size_t)n1 * 32);
  float* X2 = fbuf(P, (size_t)n2 * 64);
  float* X3 = fbuf(P, (size_t)n3 * 128);
  float* Z0 = fbuf(P, (size_t)N * 32);
  float* Z1 = fbuf(P, (size_t)N * 256);
  float* Z2 = fbuf(P, (size_t)N * 128);

  size_t amax = 0;
  auto mx = [&](size_t v) { if (v > amax) amax = v; };
  mx((size_t)n0 * 36);
  mx((size_t)n0 * 32 + (size_t)n1 * 128);
  mx((size_t)n2 * 224);
  mx((size_t)n3 * 448);
  mx((size_t)n4 * 1152 + (size_t)N * 256);
  mx((size_t)n4 * 256 + (size_t)n3 * 1416);
  mx((size_t)n2 * 712);
  mx((size_t)N * 128);
  mx((size_t)n2 * 128 + (size_t)n1 * 520);
  mx((size_t)n0 * 520);
  mx((size_t)N * 192);
  const size_t arenaBytes = (amax + 8192) * sizeof(float);
  Bump AR[2] = { Bump{ (char*)P.take(arenaBytes), 0 },
                 Bump{ (char*)P.take(arenaBytes), 0 } };
  int cur = 1;
  auto phase = [&]() -> Bump& { cur ^= 1; AR[cur].off = 0; return AR[cur]; };

  // ------------------------- launch helpers --------------------------------
  auto ew = [](long long n) { return dim3((unsigned)((n + 255) / 256)); };
  auto g2 = [](int C, int rows) { return dim3((unsigned)((C + 63) / 64), (unsigned)rows); };

  auto gg = [&](const float* X, int cin, const int* idx, int K,
                const unsigned short* W, const float* bias, const BNp* bn,
                const float* res, int res_ld, float* out, int out_ld,
                int M, int cout, int relu) {
    dim3 grid((M + 15) / 16, (cout + 63) / 64);
    const float* g_ = bn ? bn->g : nullptr;
    const float* b_ = bn ? bn->b : nullptr;
    const float* m_ = bn ? bn->m : nullptr;
    const float* v_ = bn ? bn->v : nullptr;
    if ((cin & 31) == 0)
      gg_wmma<true><<<grid, 128, 0, stream>>>(X, cin, idx, K, W, bias,
          g_, b_, m_, v_, res, res_ld, out, out_ld, M, cout, relu);
    else
      gg_wmma<false><<<grid, 128, 0, stream>>>(X, cin, idx, K, W, bias,
          g_, b_, m_, v_, res, res_ld, out, out_ld, M, cout, relu);
  };

  auto p2v = [&](const float* z, int C, const int* pidx, const float* cnt,
                 int nv, float* vox) {
    const long long tv = (long long)nv * C;
    k_fill0<<<ew(tv), 256, 0, stream>>>(vox, tv);
    k_p2v_scatter<<<g2(C, N), 64, 0, stream>>>(z, pidx, vox, C);
    k_rows_scale<<<g2(C, nv), 64, 0, stream>>>(vox, cnt, C);
  };

  auto v2p = [&](const float* vox, int C, const int* idx8, const float* w8,
                 const float* add, float* out) {
    k_v2p<<<g2(C, N), 64, 0, stream>>>(vox, idx8, w8, add, out, C);
  };

  auto resb = [&](const float* x, int ci, const int* nbr, const RBp& rb,
                  int n, int co, float* out, float* t1, float* sb) {
    gg(x, ci, nbr, 27, rb.w1, nullptr, &rb.bn1, nullptr, 0, t1, co, n, co, 1);
    const float* s = x;                         // identity shortcut (ci == co)
    if (rb.wd) {
      gg(x, ci, nullptr, 1, rb.wd, nullptr, &rb.bnd, nullptr, 0, sb, co, n, co, 0);
      s = sb;
    }
    gg(t1, co, nbr, 27, rb.w2, nullptr, &rb.bn2, s, co, out, co, n, co, 1);
  };

  auto up = [&](Bump& S, const float* x, int cx, const float* skip, int cskip,
                const Upp& U, int cy, const int* q, const int* kk,
                const int* nbr, int n) -> float* {
    int* tidx = ibuf(S, (size_t)n * 8);
    k_build_tidx<<<ew(n), 256, 0, stream>>>(q, kk, tidx, n);
    const int cc = cy + cskip;
    float* cat = fbuf(S, (size_t)n * cc);
    gg(x, cx, tidx, 8, U.w, nullptr, &U.bn, nullptr, 0, cat, cc, n, cy, 1);
    k_copy_cols<<<g2(cskip, n), 64, 0, stream>>>(skip, cskip, cat + cy, cc, cskip);
    float* t1  = fbuf(S, (size_t)n * cy);
    float* sb  = fbuf(S, (size_t)n * cy);
    float* y0  = fbuf(S, (size_t)n * cy);
    float* out = fbuf(S, (size_t)n * cy);
    resb(cat, cc, nbr, U.r0, n, cy, y0, t1, sb);
    resb(y0, cy, nbr, U.r1, n, cy, out, t1, sb);
    return out;
  };

  // ------------------------------ forward ----------------------------------
  { // stem -> X0, Z0
    Bump& S = phase();
    float* pvf = fbuf(S, (size_t)n0 * 4);
    float* s1  = fbuf(S, (size_t)n0 * 32);
    p2v(feat, 4, pv0_idx, pv0_cnt, n0, pvf);
    gg(pvf, 4, nbr0, 27, stem_w1, nullptr, &stem_bn1, nullptr, 0, s1, 32, n0, 32, 1);
    gg(s1, 32, nbr0, 27, stem_w2, nullptr, &stem_bn2, nullptr, 0, X0, 32, n0, 32, 1);
    v2p(X0, 32, dv0_idx, dv0_w, nullptr, Z0);
  }
  { // stage1 -> X1
    Bump& S = phase();
    float* v1 = fbuf(S, (size_t)n0 * 32);
    float* d  = fbuf(S, (size_t)n1 * 32);
    float* t1 = fbuf(S, (size_t)n1 * 32);
    float* sb = fbuf(S, (size_t)n1 * 32);
    float* y0 = fbuf(S, (size_t)n1 * 32);
    p2v(Z0, 32, pv0_idx, pv0_cnt, n0, v1);
    gg(v1, 32, down1, 8, st1.dw, nullptr, &st1.dbn, nullptr, 0, d, 32, n1, 32, 1);
    resb(d, 32, nbr1, st1.r0, n1, 32, y0, t1, sb);
    resb(y0, 32, nbr1, st1.r1, n1, 32, X1, t1, sb);
  }
  { // stage2 -> X2
    Bump& S = phase();
    float* d  = fbuf(S, (size_t)n2 * 32);
    float* t1 = fbuf(S, (size_t)n2 * 64);
    float* sb = fbuf(S, (size_t)n2 * 64);
    float* y0 = fbuf(S, (size_t)n2 * 64);
    gg(X1, 32, down2, 8, st2.dw, nullptr, &st2.dbn, nullptr, 0, d, 32, n2, 32, 1);
    resb(d, 32, nbr2, st2.r0, n2, 64, y0, t1, sb);
    resb(y0, 64, nbr2, st2.r1, n2, 64, X2, t1, sb);
  }
  { // stage3 -> X3
    Bump& S = phase();
    float* d  = fbuf(S, (size_t)n3 * 64);
    float* t1 = fbuf(S, (size_t)n3 * 128);
    float* sb = fbuf(S, (size_t)n3 * 128);
    float* y0 = fbuf(S, (size_t)n3 * 128);
    gg(X2, 64, down3, 8, st3.dw, nullptr, &st3.dbn, nullptr, 0, d, 64, n3, 64, 1);
    resb(d, 64, nbr3, st3.r0, n3, 128, y0, t1, sb);
    resb(y0, 128, nbr3, st3.r1, n3, 128, X3, t1, sb);
  }
  { // stage4 + point-MLP0 -> Z1
    Bump& S = phase();
    float* d   = fbuf(S, (size_t)n4 * 128);
    float* t1  = fbuf(S, (size_t)n4 * 256);
    float* sb  = fbuf(S, (size_t)n4 * 256);
    float* y0  = fbuf(S, (size_t)n4 * 256);
    float* x4  = fbuf(S, (size_t)n4 * 256);
    float* m0b = fbuf(S, (size_t)N * 256);
    gg(X3, 128, down4, 8, st4.dw, nullptr, &st4.dbn, nullptr, 0, d, 128, n4, 128, 1);
    resb(d, 128, nbr4, st4.r0, n4, 256, y0, t1, sb);
    resb(y0, 256, nbr4, st4.r1, n4, 256, x4, t1, sb);
    gg(Z0, 32, nullptr, 1, pt0.w, pt0.b, &pt0.bn, nullptr, 0, m0b, 256, N, 256, 1);
    v2p(x4, 256, dv4_idx, dv4_w, m0b, Z1);
  }
  float* y1;
  { // up1 (decoder level 3)
    Bump& S = phase();
    float* v = fbuf(S, (size_t)n4 * 256);
    p2v(Z1, 256, pv4_idx, pv4_cnt, n4, v);
    y1 = up(S, v, 256, X3, 128, u1, 256, up3_q, up3_k, nbr3, n3);
  }
  float* y2;
  { // up2 (decoder level 2)
    Bump& S = phase();
    y2 = up(S, y1, 256, X2, 64, u2, 128, up2_q, up2_k, nbr2, n2);
  }
  { // z2 = v2p(y2) + MLP1(z1)  -> Z2
    Bump& S = phase();
    float* m1b = fbuf(S, (size_t)N * 128);
    gg(Z1, 256, nullptr, 1, pt1.w, pt1.b, &pt1.bn, nullptr, 0, m1b, 128, N, 128, 1);
    v2p(y2, 128, dv2_idx, dv2_w, m1b, Z2);
  }
  float* y3;
  { // up3 (decoder level 1)
    Bump& S = phase();
    float* v = fbuf(S, (size_t)n2 * 128);
    p2v(Z2, 128, pv2_idx, pv2_cnt, n2, v);
    y3 = up(S, v, 128, X1, 32, u3, 96, up1_q, up1_k, nbr1, n1);
  }
  float* y4;
  { // up4 (decoder level 0)
    Bump& S = phase();
    y4 = up(S, y3, 96, X0, 32, u4, 96, up0_q, up0_k, nbr0, n0);
  }
  { // z3 = v2p(y4) + MLP2(z2); logits -> d_out
    Bump& S = phase();
    float* m2b = fbuf(S, (size_t)N * 96);
    float* z3  = fbuf(S, (size_t)N * 96);
    gg(Z2, 128, nullptr, 1, pt2.w, pt2.b, &pt2.bn, nullptr, 0, m2b, 96, N, 96, 1);
    v2p(y4, 96, dv0_idx, dv0_w, m2b, z3);
    gg(z3, 96, nullptr, 1, cls_w, cls_b, nullptr, nullptr, 0,
       (float*)d_out, 19, N, 19, 0);
  }
}
